// DrBC_89154931130446
// MI455X (gfx1250) — matrix operand order repaired and checked by reading, compiled, not run on hardware
//
#include <hip/hip_runtime.h>
#include <hip/hip_bf16.h>

// ---------------------------------------------------------------------------
// Types for CDNA5 WMMA (wave32): A/B = 16 bf16 per lane, C/D = 8 f32 per lane
// ---------------------------------------------------------------------------
typedef __attribute__((ext_vector_type(16))) __bf16 v16bf;
typedef __attribute__((ext_vector_type(8)))  float  v8f;

union FragAB { v16bf v; uint4 q[2]; };
union FragC  { v8f   v; float f[8]; };

__device__ __forceinline__ v8f wmma_bf16(v16bf a, v16bf b, v8f c) {
    // D = A(16x32 bf16) * B(32x16 bf16) + C(16x16 f32)
    return __builtin_amdgcn_wmma_f32_16x16x32_bf16(
        /*neg_a=*/false, a, /*neg_b=*/false, b,
        /*c_mod=*/(short)0, c, /*reuse_a=*/false, /*reuse_b=*/false);
}

// A fragment: rows = row0..row0+15 (M), K = kb..kb+31. Row-major bf16, stride 128.
// Contiguous-k layout (same as B): lane<16 -> M=lane, elems[0..15]=K kb..kb+15
//             lane>=16 -> M=lane-16, elems = K kb+16..kb+31
__device__ __forceinline__ v16bf load_fragA(const __hip_bfloat16* M, int row0,
                                            int kb, int half, int lr) {
    const __hip_bfloat16* base = M + (size_t)(row0 + lr) * 128;
    int k0 = kb + half * 16;
    FragAB f;
    f.q[0] = *reinterpret_cast<const uint4*>(base + k0);
    f.q[1] = *reinterpret_cast<const uint4*>(base + k0 + 8);
    return f.v;
}

// B fragment: B[k][n] = W[col0+n][k] (W row-major (cols x 128)).
// Contiguous-k layout (same as A): lane<16 -> N=lane, elems = K kb..kb+15; lane>=16 -> K kb+16..kb+31
__device__ __forceinline__ v16bf load_fragB(const __hip_bfloat16* W, int col0,
                                            int kb, int half, int lr) {
    const __hip_bfloat16* p = W + (size_t)(col0 + lr) * 128 + kb + half * 16;
    FragAB f;
    f.q[0] = reinterpret_cast<const uint4*>(p)[0];
    f.q[1] = reinterpret_cast<const uint4*>(p)[1];
    return f.v;
}

__device__ __forceinline__ float sigmoidf_(float x) {
    return 1.0f / (1.0f + __expf(-x));
}

// ---------------------------------------------------------------------------
// Utility kernels
// ---------------------------------------------------------------------------
__global__ void fill_zero_kernel(float* p, int n) {
    int i = blockIdx.x * blockDim.x + threadIdx.x;
    if (i < n) p[i] = 0.0f;
}

__global__ void cvt_bf16_kernel(const float* in, __hip_bfloat16* out, int n) {
    int i = blockIdx.x * blockDim.x + threadIdx.x;
    if (i < n) out[i] = __float2bfloat16(in[i]);
}

// ---------------------------------------------------------------------------
// Encoder: t[n][d] = sum_k x[n][k]*W_enc[d][k]; accumulate BN stats.
// blockDim = 128 (d), 64 rows per block.
// ---------------------------------------------------------------------------
__global__ void enc_pre_kernel(const float* __restrict__ x,
                               const float* __restrict__ W_enc,
                               float* __restrict__ t,
                               float* __restrict__ sum, float* __restrict__ sumsq,
                               int N) {
    int d = threadIdx.x;
    int base = blockIdx.x * 64;
    float w0 = W_enc[d * 3 + 0], w1 = W_enc[d * 3 + 1], w2 = W_enc[d * 3 + 2];
    float ls = 0.0f, lq = 0.0f;
    for (int r = 0; r < 64; ++r) {
        int n = base + r;
        if (n >= N) break;
        float v = x[n * 3 + 0] * w0 + x[n * 3 + 1] * w1 + x[n * 3 + 2] * w2;
        t[(size_t)n * 128 + d] = v;
        ls += v; lq += v * v;
    }
    atomicAdd(&sum[d], ls);
    atomicAdd(&sumsq[d], lq);
}

__global__ void bn_finalize_kernel(const float* sum, const float* sumsq,
                                   const float* gamma, const float* beta,
                                   float invN, float* scale, float* shift, int C) {
    int d = threadIdx.x;
    if (d >= C) return;
    float mu  = sum[d] * invN;
    float var = sumsq[d] * invN - mu * mu;
    float sc  = gamma[d] * rsqrtf(var + 1e-5f);
    scale[d] = sc;
    shift[d] = beta[d] - mu * sc;
}

// Apply encoder BN + relu; init h, h_sum (pad rows -> 0)
__global__ void enc_apply_kernel(float* __restrict__ h, float* __restrict__ hsum,
                                 const float* __restrict__ scale,
                                 const float* __restrict__ shift,
                                 int N, int total) {
    int i = blockIdx.x * blockDim.x + threadIdx.x;
    if (i >= total) return;
    int n = i >> 7, d = i & 127;
    float v = 0.0f;
    if (n < N) v = fmaxf(h[i] * scale[d] + shift[d], 0.0f);
    h[i] = v;
    hsum[i] = v;
}

// ---------------------------------------------------------------------------
// Edge scatter: one wave per edge; lane handles 4 columns (float4).
// h_agg[dst] += norm * h[src]   (global f32 atomics, L2-resident)
// ---------------------------------------------------------------------------
__global__ void scatter_kernel(const int* __restrict__ src,
                               const int* __restrict__ dst,
                               const float* __restrict__ norm,
                               const float* __restrict__ h,
                               float* __restrict__ hagg, int E) {
    int e = blockIdx.x * (blockDim.x >> 5) + (threadIdx.x >> 5);
    int lane = threadIdx.x & 31;
    if (e >= E) return;
    int s = src[e], d = dst[e];
    float w = norm[e];
    float4 v = reinterpret_cast<const float4*>(h + (size_t)s * 128)[lane];
    float* out = hagg + (size_t)d * 128 + lane * 4;
    atomicAdd(out + 0, w * v.x);
    atomicAdd(out + 1, w * v.y);
    atomicAdd(out + 2, w * v.z);
    atomicAdd(out + 3, w * v.w);
}

// h_sum += h_agg ; snapshot h_agg and h to bf16 for the WMMA GEMMs
__global__ void post_scatter_kernel(const float* __restrict__ hagg,
                                    const float* __restrict__ h,
                                    float* __restrict__ hsum,
                                    __hip_bfloat16* __restrict__ ha_bf,
                                    __hip_bfloat16* __restrict__ h_bf,
                                    int total) {
    int i = blockIdx.x * blockDim.x + threadIdx.x;
    if (i >= total) return;
    float a = hagg[i];
    hsum[i] += a;
    ha_bf[i] = __float2bfloat16(a);
    h_bf[i]  = __float2bfloat16(h[i]);
}

// ---------------------------------------------------------------------------
// GRU via WMMA bf16: one wave (32 threads) per 16-row tile.
// gi = ha @ W_ih^T, gh = h @ W_hh^T (each N x 384); gates fused in epilogue.
// ---------------------------------------------------------------------------
__global__ void __launch_bounds__(32)
gru_kernel(const __hip_bfloat16* __restrict__ ha_bf,
           const __hip_bfloat16* __restrict__ h_bf,
           const __hip_bfloat16* __restrict__ Wih,
           const __hip_bfloat16* __restrict__ Whh,
           const float* __restrict__ b_ih, const float* __restrict__ b_hh,
           float* __restrict__ h) {
    int rt   = blockIdx.x;
    int lane = threadIdx.x & 31;
    int half = lane >> 4;
    int lr   = lane & 15;
    int row0 = rt * 16;

    for (int jt = 0; jt < 8; ++jt) {       // 8 col tiles of 16 -> 128 outputs
        FragC gi_r = {}, gh_r = {}, gi_z = {}, gh_z = {}, gi_n = {}, gh_n = {};
        int j0 = jt * 16;
#pragma unroll
        for (int kk = 0; kk < 4; ++kk) {   // K = 128 in steps of 32
            int kb = kk * 32;
            v16bf a_i = load_fragA(ha_bf, row0, kb, half, lr);
            v16bf a_h = load_fragA(h_bf,  row0, kb, half, lr);
            gi_r.v = wmma_bf16(a_i, load_fragB(Wih, j0,        kb, half, lr), gi_r.v);
            gi_z.v = wmma_bf16(a_i, load_fragB(Wih, 128 + j0,  kb, half, lr), gi_z.v);
            gi_n.v = wmma_bf16(a_i, load_fragB(Wih, 256 + j0,  kb, half, lr), gi_n.v);
            gh_r.v = wmma_bf16(a_h, load_fragB(Whh, j0,        kb, half, lr), gh_r.v);
            gh_z.v = wmma_bf16(a_h, load_fragB(Whh, 128 + j0,  kb, half, lr), gh_z.v);
            gh_n.v = wmma_bf16(a_h, load_fragB(Whh, 256 + j0,  kb, half, lr), gh_n.v);
        }
        int col = j0 + lr;                 // C/D layout: N = lane&15 fixed per lane
        float bir = b_ih[col],       bhr = b_hh[col];
        float biz = b_ih[128 + col], bhz = b_hh[128 + col];
        float bin = b_ih[256 + col], bhn = b_hh[256 + col];
#pragma unroll
        for (int i = 0; i < 8; ++i) {      // rows M = i + 8*half
            int row = row0 + i + 8 * half;
            float r  = sigmoidf_(gi_r.f[i] + bir + gh_r.f[i] + bhr);
            float zz = sigmoidf_(gi_z.f[i] + biz + gh_z.f[i] + bhz);
            float nn = tanhf(gi_n.f[i] + bin + r * (gh_n.f[i] + bhn));
            size_t idx = (size_t)row * 128 + col;
            float hold = h[idx];
            h[idx] = (1.0f - zz) * nn + zz * hold;
        }
    }
}

// z_bf = bf16(h_sum / 6)
__global__ void make_z_kernel(const float* __restrict__ hsum,
                              __hip_bfloat16* __restrict__ z_bf, int total) {
    int i = blockIdx.x * blockDim.x + threadIdx.x;
    if (i >= total) return;
    z_bf[i] = __float2bfloat16(hsum[i] * (1.0f / 6.0f));
}

// Decoder GEMM: t = z @ W_dec^T (N x 64) via WMMA + BN stat accumulation.
__global__ void __launch_bounds__(32)
dec_gemm_kernel(const __hip_bfloat16* __restrict__ z_bf,
                const __hip_bfloat16* __restrict__ Wdec,
                float* __restrict__ t,
                float* __restrict__ dsum, float* __restrict__ dsumsq) {
    int rt   = blockIdx.x;
    int lane = threadIdx.x & 31;
    int half = lane >> 4;
    int lr   = lane & 15;
    int row0 = rt * 16;

    for (int jt = 0; jt < 4; ++jt) {       // 64 output cols
        FragC acc = {};
#pragma unroll
        for (int kk = 0; kk < 4; ++kk) {
            int kb = kk * 32;
            v16bf a = load_fragA(z_bf, row0, kb, half, lr);
            v16bf b = load_fragB(Wdec, jt * 16, kb, half, lr);
            acc.v = wmma_bf16(a, b, acc.v);
        }
        int col = jt * 16 + lr;
        float ls = 0.0f, lq = 0.0f;
#pragma unroll
        for (int i = 0; i < 8; ++i) {
            int row = row0 + i + 8 * half;
            float v = acc.f[i];
            t[(size_t)row * 64 + col] = v;
            ls += v; lq += v * v;
        }
        atomicAdd(&dsum[col], ls);
        atomicAdd(&dsumsq[col], lq);
    }
}

// y[n] = sum_j relu(BN(t[n][j])) * W_dec2[j]
__global__ void dec_out_kernel(const float* __restrict__ t,
                               const float* __restrict__ dscale,
                               const float* __restrict__ dshift,
                               const float* __restrict__ Wdec2,
                               float* __restrict__ y, int N) {
    int n = blockIdx.x * blockDim.x + threadIdx.x;
    if (n >= N) return;
    const float* tn = t + (size_t)n * 64;
    float acc = 0.0f;
#pragma unroll 8
    for (int j = 0; j < 64; ++j) {
        float v = fmaxf(tn[j] * dscale[j] + dshift[j], 0.0f);
        acc += v * Wdec2[j];
    }
    y[n] = acc;
}

// ---------------------------------------------------------------------------
// Host-side orchestration (graph-capture safe: launches only)
// ---------------------------------------------------------------------------
extern "C" void kernel_launch(void* const* d_in, const int* in_sizes, int n_in,
                              void* d_out, int out_size, void* d_ws, size_t ws_size,
                              hipStream_t stream) {
    const float* x      = (const float*)d_in[0];
    const int*   eidx   = (const int*)d_in[1];
    const float* norm   = (const float*)d_in[2];
    const float* W_enc  = (const float*)d_in[3];
    const float* g_e    = (const float*)d_in[4];
    const float* b_e    = (const float*)d_in[5];
    const float* W_ih   = (const float*)d_in[6];
    const float* W_hh   = (const float*)d_in[7];
    const float* b_ih   = (const float*)d_in[8];
    const float* b_hh   = (const float*)d_in[9];
    const float* W_dec  = (const float*)d_in[10];
    const float* g_d    = (const float*)d_in[11];
    const float* b_d    = (const float*)d_in[12];
    const float* W_dec2 = (const float*)d_in[13];
    float* y = (float*)d_out;

    const int N = in_sizes[0] / 3;
    const int E = in_sizes[2];
    const int* src = eidx;
    const int* dst = eidx + E;

    const size_t Npad   = ((size_t)N + 15) & ~(size_t)15;
    const size_t ND     = Npad * 128;
    const int    totalI = (int)ND;
    const int    tiles  = (int)(Npad / 16);

    // ---- workspace carve-up (256B aligned) ----
    char* base = (char*)d_ws;
    size_t off = 0;
    auto carve = [&](size_t bytes) -> char* {
        char* p = base + off;
        off = (off + bytes + 255) & ~(size_t)255;
        return p;
    };
    float* h      = (float*)carve(ND * sizeof(float));
    float* hagg   = (float*)carve(ND * sizeof(float));
    float* hsum   = (float*)carve(ND * sizeof(float));
    __hip_bfloat16* ha_bf = (__hip_bfloat16*)carve(ND * sizeof(__hip_bfloat16));
    __hip_bfloat16* h_bf  = (__hip_bfloat16*)carve(ND * sizeof(__hip_bfloat16));
    __hip_bfloat16* Wih_bf  = (__hip_bfloat16*)carve(384 * 128 * sizeof(__hip_bfloat16));
    __hip_bfloat16* Whh_bf  = (__hip_bfloat16*)carve(384 * 128 * sizeof(__hip_bfloat16));
    __hip_bfloat16* Wdec_bf = (__hip_bfloat16*)carve(64 * 128 * sizeof(__hip_bfloat16));
    float* stats = (float*)carve(1024 * sizeof(float));
    // stats layout
    float* e_sum   = stats + 0;    float* e_sumsq = stats + 128;
    float* e_scale = stats + 256;  float* e_shift = stats + 384;
    float* d_sum   = stats + 512;  float* d_sumsq = stats + 576;
    float* d_scale = stats + 640;  float* d_shift = stats + 704;
    // buffer reuse: z_bf overlays ha_bf; t (N x 64 f32) overlays h_bf region
    __hip_bfloat16* z_bf = ha_bf;
    float* t = (float*)h_bf;

    const int TB = 256;
    const int ew_blocks = (totalI + TB - 1) / TB;

    // 0) zero BN stats accumulators
    fill_zero_kernel<<<(1024 + TB - 1) / TB, TB, 0, stream>>>(stats, 1024);

    // 1) weights -> bf16 (once per call)
    cvt_bf16_kernel<<<(384 * 128 + TB - 1) / TB, TB, 0, stream>>>(W_ih, Wih_bf, 384 * 128);
    cvt_bf16_kernel<<<(384 * 128 + TB - 1) / TB, TB, 0, stream>>>(W_hh, Whh_bf, 384 * 128);
    cvt_bf16_kernel<<<(64 * 128 + TB - 1) / TB, TB, 0, stream>>>(W_dec, Wdec_bf, 64 * 128);

    // 2) encoder + batchnorm + relu
    enc_pre_kernel<<<(N + 63) / 64, 128, 0, stream>>>(x, W_enc, h, e_sum, e_sumsq, N);
    bn_finalize_kernel<<<1, 128, 0, stream>>>(e_sum, e_sumsq, g_e, b_e,
                                              1.0f / (float)N, e_scale, e_shift, 128);
    enc_apply_kernel<<<ew_blocks, TB, 0, stream>>>(h, hsum, e_scale, e_shift, N, totalI);

    // 3) propagation layers
    const int sc_blocks = (E + 7) / 8;
    for (int l = 0; l < 5; ++l) {
        fill_zero_kernel<<<ew_blocks, TB, 0, stream>>>(hagg, totalI);
        scatter_kernel<<<sc_blocks, 256, 0, stream>>>(src, dst, norm, h, hagg, E);
        post_scatter_kernel<<<ew_blocks, TB, 0, stream>>>(hagg, h, hsum, ha_bf, h_bf, totalI);
        if (l < 4) {  // layer-5 GRU output is dead (only h_sum is consumed)
            gru_kernel<<<tiles, 32, 0, stream>>>(ha_bf, h_bf, Wih_bf, Whh_bf,
                                                 b_ih, b_hh, h);
        }
    }

    // 4) decoder
    make_z_kernel<<<ew_blocks, TB, 0, stream>>>(hsum, z_bf, totalI);
    dec_gemm_kernel<<<tiles, 32, 0, stream>>>(z_bf, Wdec_bf, t, d_sum, d_sumsq);
    bn_finalize_kernel<<<1, 128, 0, stream>>>(d_sum, d_sumsq, g_d, b_d,
                                              1.0f / (float)N, d_scale, d_shift, 64);
    dec_out_kernel<<<(N + TB - 1) / TB, TB, 0, stream>>>(t, d_scale, d_shift, W_dec2, y, N);
}
